// MDLSTM_hwr_80676665688422
// MI455X (gfx1250) — matrix operand order repaired and checked by reading, not compiled, not run on hardware
//
#include <hip/hip_runtime.h>
#include <math.h>

typedef _Float16 v16h __attribute__((ext_vector_type(16)));
typedef _Float16 v8h  __attribute__((ext_vector_type(8)));
typedef float    v8f  __attribute__((ext_vector_type(8)));

__device__ __forceinline__ v16h cat8(v8h a, v8h b) {
  return __builtin_shufflevector(a, b, 0, 1, 2, 3, 4, 5, 6, 7, 8, 9, 10, 11, 12, 13, 14, 15);
}

__device__ __forceinline__ float sigmoidf_(float x) { return 1.0f / (1.0f + expf(-x)); }

static inline int ceil16(int v) { return (v + 15) & ~15; }
static inline int ceil32(int v) { return (v + 31) & ~31; }

// ---------------------------------------------------------------------------
// Pack conv weights [Cout x Cin*9] f32 -> zero-padded [Mp x Kp] f16 row-major.
// ---------------------------------------------------------------------------
__global__ void pack_conv_A(const float* __restrict__ w, _Float16* __restrict__ A,
                            int M, int Kdim, int Mp, int Kp) {
  int idx = blockIdx.x * blockDim.x + threadIdx.x;
  if (idx >= Mp * Kp) return;
  int m = idx / Kp, k = idx - m * Kp;
  A[idx] = (m < M && k < Kdim) ? (_Float16)w[m * Kdim + k] : (_Float16)0.0f;
}

// ---------------------------------------------------------------------------
// Pack an im2col N-chunk, f16 COLUMN-major: Bc[(n - n0)*Kp + k].
// Covers nLocal in [0, nCols), k in [0, Kp); zero pad outside tensor.
// ---------------------------------------------------------------------------
__global__ void pack_conv_B(const float* __restrict__ x, _Float16* __restrict__ Bc,
                            int B, int Cin, int Hin, int Win, int Hout, int Wout,
                            int Kdim, int Kp, int n0, int nCols, int Ntot) {
  int idx = blockIdx.x * blockDim.x + threadIdx.x;
  if (idx >= nCols * Kp) return;
  int nl = idx / Kp, k = idx - nl * Kp;
  int n = n0 + nl;
  _Float16 v = (_Float16)0.0f;
  if (n < Ntot && k < Kdim) {
    int b = n / (Hout * Wout);
    int r = n - b * Hout * Wout;
    int oy = r / Wout;
    int ox = r - oy * Wout;
    int ci = k / 9;
    int rr = k - ci * 9;
    int ky = rr / 3, kx = rr - ky * 3;
    v = (_Float16)x[((b * Cin + ci) * Hin + (oy + ky)) * Win + (ox + kx)];
  }
  Bc[idx] = v;
}

// ---------------------------------------------------------------------------
// Packed WMMA GEMM + bias + tanh epilogue (conv). A:[Mp x Kp] rm, Bc:[cols x Kp] cm.
// One wave per 16x16 tile; K-loop is 4x b128 + v_wmma, no predication.
// ---------------------------------------------------------------------------
__global__ __launch_bounds__(128) void conv_gemm_wmma(
    const _Float16* __restrict__ A, const _Float16* __restrict__ Bc,
    const float* __restrict__ bias, float* __restrict__ y,
    int Kp, int Cout, int n0, int Ntot, int Hout, int Wout,
    int tilesM, int tilesN) {
  const int tile = blockIdx.x * (blockDim.x >> 5) + (threadIdx.x >> 5);
  if (tile >= tilesM * tilesN) return;
  const int tm = tile % tilesM;
  const int tn = tile / tilesM;
  const int lane = threadIdx.x & 31;
  const int laneLo = lane & 15;
  const int laneHi = lane >> 4;

  const _Float16* arow = A + (size_t)(tm * 16 + laneLo) * Kp;   // A row (M = laneLo)
  const _Float16* bcol = Bc + (size_t)(tn * 16 + laneLo) * Kp;  // B column (N = laneLo)

  v8f acc = {};
  for (int k0 = 0; k0 < Kp; k0 += 32) {
    v8h a0 = *(const v8h*)(arow + k0 + 8 * laneHi);        // A: K = j + 8*(j>=8 + laneHi)
    v8h a1 = *(const v8h*)(arow + k0 + 16 + 8 * laneHi);
    v8h b0 = *(const v8h*)(bcol + k0 + 8 * laneHi);        // B: K = j + 8*(j>=8 + laneHi)
    v8h b1 = *(const v8h*)(bcol + k0 + 16 + 8 * laneHi);
    acc = __builtin_amdgcn_wmma_f32_16x16x32_f16(false, cat8(a0, a1), false, cat8(b0, b1),
                                                 (short)0, acc, false, false);
  }

  const int n = n0 + tn * 16 + laneLo;
  if (n >= Ntot) return;
  int b = n / (Hout * Wout);
  int r = n - b * Hout * Wout;
  int oy = r / Wout;
  int ox = r - oy * Wout;
#pragma unroll
  for (int rr = 0; rr < 8; ++rr) {
    int m = tm * 16 + rr + 8 * laneHi;  // D layout: lane N, VGPR rr -> M = rr + 8*laneHi
    if (m < Cout)
      y[((b * Cout + m) * Hout + oy) * Wout + ox] = tanhf(acc[rr] + bias[m]);
  }
}

// ---------------------------------------------------------------------------
// Max pool (kh,kw) stride (kh,kw), VALID.
// ---------------------------------------------------------------------------
__global__ void maxpool_kernel(const float* __restrict__ x, float* __restrict__ y,
                               int B, int C, int H, int W, int kh, int kw,
                               int Ho, int Wo) {
  int idx = blockIdx.x * blockDim.x + threadIdx.x;
  int total = B * C * Ho * Wo;
  if (idx >= total) return;
  int ox = idx % Wo;
  int t = idx / Wo;
  int oy = t % Ho; t /= Ho;
  int c = t % C;
  int b = t / C;
  float m = -3.402823466e38f;
  for (int r = 0; r < kh; ++r)
    for (int s = 0; s < kw; ++s)
      m = fmaxf(m, x[((b * C + c) * H + (oy * kh + r)) * W + (ox * kw + s)]);
  y[idx] = m;
}

// ---------------------------------------------------------------------------
// Fused LSTM A: [Wis | Wss[...,0] | Wss[...,1]] -> [M x Kp] f16 (M=4h).
// ---------------------------------------------------------------------------
__global__ void pack_lstm_A(const float* __restrict__ Wis, const float* __restrict__ Wss,
                            _Float16* __restrict__ A, int M, int C, int hid, int Kp) {
  int idx = blockIdx.x * blockDim.x + threadIdx.x;
  if (idx >= M * Kp) return;
  int m = idx / Kp, k = idx - m * Kp;
  int Kdim = C + 2 * hid;
  float v = 0.0f;
  if (k < C)                 v = Wis[m * C + k];
  else if (k < C + hid)      v = Wss[(m * hid + (k - C)) * 2 + 0];        // W_up
  else if (k < Kdim)         v = Wss[(m * hid + (k - C - hid)) * 2 + 1];  // W_cur
  A[idx] = (_Float16)v;
}

// ---------------------------------------------------------------------------
// Per-diagonal B panel, f16 column-major [N x Kp]: [x_diag ; h_up ; h_cur].
// N = B*H (multiple of 16 for all layers here).
// ---------------------------------------------------------------------------
__global__ void pack_lstm_B(const float* __restrict__ x, const float* __restrict__ hprev,
                            _Float16* __restrict__ Bc,
                            int B, int C, int H, int W, int hid, int diag, int Kp) {
  int N = B * H;
  int idx = blockIdx.x * blockDim.x + threadIdx.x;
  if (idx >= N * Kp) return;
  int n = idx / Kp, k = idx - n * Kp;
  int b = n / H;
  int i = n - b * H;
  int j = diag - i;
  float v = 0.0f;
  if (k < C) {
    if (j >= 0 && j < W) v = x[((b * C + k) * H + i) * W + j];
  } else if (k < C + hid) {
    if (i > 0) v = hprev[(b * hid + (k - C)) * H + (i - 1)];
  } else if (k < C + 2 * hid) {
    v = hprev[(b * hid + (k - C - hid)) * H + i];
  }
  Bc[idx] = (_Float16)v;
}

// ---------------------------------------------------------------------------
// Packed WMMA GEMM for the diagonal step: g[M x N] = A @ B.
// ---------------------------------------------------------------------------
__global__ __launch_bounds__(128) void lstm_gemm_wmma(
    const _Float16* __restrict__ A, const _Float16* __restrict__ Bc,
    float* __restrict__ g, int Kp, int M, int N, int tilesM, int tilesN) {
  const int tile = blockIdx.x * (blockDim.x >> 5) + (threadIdx.x >> 5);
  if (tile >= tilesM * tilesN) return;
  const int tm = tile % tilesM;
  const int tn = tile / tilesM;
  const int lane = threadIdx.x & 31;
  const int laneLo = lane & 15;
  const int laneHi = lane >> 4;

  const _Float16* arow = A + (size_t)(tm * 16 + laneLo) * Kp;
  const _Float16* bcol = Bc + (size_t)(tn * 16 + laneLo) * Kp;

  v8f acc = {};
  for (int k0 = 0; k0 < Kp; k0 += 32) {
    v8h a0 = *(const v8h*)(arow + k0 + 8 * laneHi);
    v8h a1 = *(const v8h*)(arow + k0 + 16 + 8 * laneHi);
    v8h b0 = *(const v8h*)(bcol + k0 + 8 * laneHi);
    v8h b1 = *(const v8h*)(bcol + k0 + 16 + 8 * laneHi);
    acc = __builtin_amdgcn_wmma_f32_16x16x32_f16(false, cat8(a0, a1), false, cat8(b0, b1),
                                                 (short)0, acc, false, false);
  }

  const int n = tn * 16 + laneLo;  // N is a multiple of 16
#pragma unroll
  for (int rr = 0; rr < 8; ++rr) {
    int m = tm * 16 + rr + 8 * laneHi;
    g[m * N + n] = acc[rr];
  }
}

// ---------------------------------------------------------------------------
// Gates, cell/hidden update, diagonal mask, scatter into un-skewed output.
// ---------------------------------------------------------------------------
__global__ void lstm_diag_gates(const float* __restrict__ g, const float* __restrict__ bis,
                                float* __restrict__ h, float* __restrict__ c,
                                float* __restrict__ out /*[B,h,H,W]*/,
                                int B, int H, int W, int hid, int diag) {
  int idx = blockIdx.x * blockDim.x + threadIdx.x;
  int total = B * hid * H;
  if (idx >= total) return;
  int i = idx % H;
  int t = idx / H;
  int jh = t % hid;
  int b = t / hid;
  int N = B * H;
  int n = b * H + i;
  float ig = g[jh * N + n]             + bis[jh];
  float fg = g[(hid + jh) * N + n]     + bis[hid + jh];
  float og = g[(2 * hid + jh) * N + n] + bis[2 * hid + jh];
  float cg = g[(3 * hid + jh) * N + n] + bis[3 * hid + jh];
  float c_old = c[(b * hid + jh) * H + i];
  float cn = sigmoidf_(fg) * c_old + sigmoidf_(ig) * tanhf(cg);
  float hn = sigmoidf_(og) * tanhf(cn);
  int j = diag - i;
  bool valid = (j >= 0) && (j < W);
  h[(b * hid + jh) * H + i] = valid ? hn : 0.0f;
  c[(b * hid + jh) * H + i] = valid ? cn : 0.0f;
  if (valid) out[((b * hid + jh) * H + i) * W + j] = hn;
}

__global__ void zero_kernel(float* __restrict__ p, int n) {
  int i = blockIdx.x * blockDim.x + threadIdx.x;
  if (i < n) p[i] = 0.0f;
}

// ---------------------------------------------------------------------------
// FC + log_softmax over the BATCH axis (dim=1, as in the reference).
// xin = [B,120,62]; out = [62,B,80].
// ---------------------------------------------------------------------------
__global__ void fc_logsoftmax(const float* __restrict__ xin,
                              const float* __restrict__ fw, const float* __restrict__ fb,
                              float* __restrict__ out, int B, int Cin, int Wd, int Cout) {
  int idx = blockIdx.x * blockDim.x + threadIdx.x;
  if (idx >= Wd * Cout) return;
  int c = idx % Cout;
  int w = idx / Cout;
  float z[16];
  float mx = -3.402823466e38f;
  for (int b = 0; b < B; ++b) {
    float s = fb[c];
    for (int ch = 0; ch < Cin; ++ch)
      s += xin[(b * Cin + ch) * Wd + w] * fw[c * Cin + ch];
    z[b] = s;
    mx = fmaxf(mx, s);
  }
  float se = 0.0f;
  for (int b = 0; b < B; ++b) se += expf(z[b] - mx);
  float lse = mx + logf(se);
  for (int b = 0; b < B; ++b)
    out[(w * B + b) * Cout + c] = z[b] - lse;
}

// ---------------------------------------------------------------------------
extern "C" void kernel_launch(void* const* d_in, const int* in_sizes, int n_in,
                              void* d_out, int out_size, void* d_ws, size_t ws_size,
                              hipStream_t stream) {
  const float* input = (const float*)d_in[0];
  const float* conv_w[4] = {(const float*)d_in[1],  (const float*)d_in[6],
                            (const float*)d_in[11], (const float*)d_in[16]};
  const float* conv_b[4] = {(const float*)d_in[2],  (const float*)d_in[7],
                            (const float*)d_in[12], (const float*)d_in[17]};
  const float* Wis[4]    = {(const float*)d_in[3],  (const float*)d_in[8],
                            (const float*)d_in[13], (const float*)d_in[18]};
  const float* bis[4]    = {(const float*)d_in[4],  (const float*)d_in[9],
                            (const float*)d_in[14], (const float*)d_in[19]};
  const float* Wss[4]    = {(const float*)d_in[5],  (const float*)d_in[10],
                            (const float*)d_in[15], (const float*)d_in[20]};
  const float* fc_w = (const float*)d_in[21];
  const float* fc_b = (const float*)d_in[22];

  // Workspace carve-up (float units). Peak ~126 MB.
  float* convbuf = (float*)d_ws;              // <= 15,207,360 (conv0 pre-pool)
  float* poolbuf = convbuf + 11000000;        // <=  3,801,840 (pool0)
  float* lstmbuf = poolbuf + 4000000;         // <=  7,603,680 (lstm0)
  float* hbuf    = lstmbuf + 8000000;         // <= 14,880
  float* cbuf    = hbuf + 16384;              // <= 14,880
  float* gbuf    = cbuf + 16384;              // <= 59,520
  _Float16* Apack = (_Float16*)(gbuf + 65536);        // <= 168,960 halves
  _Float16* Bpack = (_Float16*)(gbuf + 65536 + 131072);  // 16M halves (32 MB)
  const int PACKB_HALVES = 16 * 1024 * 1024;

  const int B = 16;
  const int Cin_[4]  = {3, 30, 60, 90};
  const int Cout_[4] = {15, 45, 75, 105};
  const int Hin_[4]  = {64, 31, 14, 6};
  const int Win_[4]  = {1024, 511, 254, 126};
  const int ph_[4]   = {2, 2, 2, 4};
  const int pw_[4]   = {2, 2, 2, 2};
  const int hid_[4]  = {30, 60, 90, 120};

  const float* layer_in = input;
  for (int l = 0; l < 4; ++l) {
    const int Hout = Hin_[l] - 2, Wout = Win_[l] - 2;
    const int Ho = Hout / ph_[l], Wo = Wout / pw_[l];

    // ---- conv as packed GEMM, chunked over output pixels ----
    const int Kdim = Cin_[l] * 9;
    const int Kp = ceil32(Kdim);
    const int Mp = ceil16(Cout_[l]);
    const int tilesM = Mp / 16;
    {
      int acnt = Mp * Kp;
      pack_conv_A<<<(acnt + 255) / 256, 256, 0, stream>>>(conv_w[l], Apack,
                                                          Cout_[l], Kdim, Mp, Kp);
    }
    const int Ntot = B * Hout * Wout;
    int Nchunk = (PACKB_HALVES / Kp) & ~15;
    if (Nchunk > ceil16(Ntot)) Nchunk = ceil16(Ntot);
    for (int n0 = 0; n0 < Ntot; n0 += Nchunk) {
      int actual = Ntot - n0;
      if (actual > Nchunk) actual = Nchunk;
      int nCols = ceil16(actual);
      int pcnt = nCols * Kp;
      pack_conv_B<<<(pcnt + 255) / 256, 256, 0, stream>>>(
          layer_in, Bpack, B, Cin_[l], Hin_[l], Win_[l], Hout, Wout,
          Kdim, Kp, n0, nCols, Ntot);
      int tilesN = nCols / 16;
      int tiles = tilesM * tilesN;
      conv_gemm_wmma<<<(tiles + 3) / 4, 128, 0, stream>>>(
          Apack, Bpack, conv_b[l], convbuf,
          Kp, Cout_[l], n0, Ntot, Hout, Wout, tilesM, tilesN);
    }

    // ---- maxpool ----
    const int ptot = B * Cout_[l] * Ho * Wo;
    maxpool_kernel<<<(ptot + 255) / 256, 256, 0, stream>>>(
        convbuf, poolbuf, B, Cout_[l], Hout, Wout, ph_[l], pw_[l], Ho, Wo);

    // ---- diagonal LSTM ----
    const int hid = hid_[l], C = Cout_[l], H = Ho, W = Wo;
    const int M = 4 * hid;                   // not a multiple of 16 for hid 30, 90
    const int KpL = ceil32(C + 2 * hid);
    const int Nl = B * H;                    // multiple of 16 for all layers
    {
      int acnt = M * KpL;
      pack_lstm_A<<<(acnt + 255) / 256, 256, 0, stream>>>(Wis[l], Wss[l], Apack,
                                                          M, C, hid, KpL);
    }
    const int st = B * hid * H;
    zero_kernel<<<(st + 255) / 256, 256, 0, stream>>>(hbuf, st);
    zero_kernel<<<(st + 255) / 256, 256, 0, stream>>>(cbuf, st);
    const int ltilesM = M / 16, ltilesN = Nl / 16;
    const int ltiles = ltilesM * ltilesN;
    const int bcnt = Nl * KpL;
    const int Kdiag = H + W - 1;
    for (int k = 0; k < Kdiag; ++k) {
      pack_lstm_B<<<(bcnt + 255) / 256, 256, 0, stream>>>(
          poolbuf, hbuf, Bpack, B, C, H, W, hid, k, KpL);
      lstm_gemm_wmma<<<(ltiles + 3) / 4, 128, 0, stream>>>(
          Apack, Bpack, gbuf, KpL, M, Nl, ltilesM, ltilesN);
      lstm_diag_gates<<<(st + 255) / 256, 256, 0, stream>>>(
          gbuf, bis[l], hbuf, cbuf, lstmbuf, B, H, W, hid, k);
    }
    layer_in = lstmbuf;
  }

  // ---- FC + log_softmax(axis=1) -> d_out [62,16,80] ----
  fc_logsoftmax<<<(62 * 80 + 255) / 256, 256, 0, stream>>>(
      lstmbuf, fc_w, fc_b, (float*)d_out, B, 120, 62, 80);
}
